// CrossModalAttention_3393024164014
// MI455X (gfx1250) — matrix operand order repaired and checked by reading, compiled, not run on hardware
//
#include <hip/hip_runtime.h>

// ---------------------------------------------------------------------------
// CrossModalAttention for MI455X (gfx1250), wave32 + WMMA bf16.
//
// Factored algorithm: energy = x^T (Wq^T k) + bq^T k  collapses the Ci=512
// dim out of the 13.2-GFLOP hot path, leaving ~1.5 GFLOP of WMMA work on a
// ~115 MB memory-bound problem (floor ~5us @ 23.3 TB/s HBM).
// ---------------------------------------------------------------------------

typedef __bf16 bf16x16 __attribute__((ext_vector_type(16)));
typedef float  f32x8   __attribute__((ext_vector_type(8)));

#define NB   16    // batch
#define CR   1024  // Cr
#define CI   512   // Ci
#define CS   256   // Cs
#define NS   25    // skeleton positions (after width-mean pool)
#define NSP  32    // NS padded to WMMA K/N granule
#define HW   784   // 28*28
#define VROW 40    // ushorts per LDS row (16B-aligned, bank-spread stride)

__device__ __forceinline__ unsigned short f2bf(float f) {
  unsigned int u = __float_as_uint(f);
  u += 0x7FFFu + ((u >> 16) & 1u);          // round-to-nearest-even
  return (unsigned short)(u >> 16);
}

// Two aligned 16B loads -> one 16-element bf16 WMMA fragment.
__device__ __forceinline__ bf16x16 load_frag16(const unsigned short* p0,
                                               const unsigned short* p1) {
  union { uint4 q[2]; bf16x16 v; } u;
  u.q[0] = *reinterpret_cast<const uint4*>(p0);
  u.q[1] = *reinterpret_cast<const uint4*>(p1);
  return u.v;
}

// --------------------------- prep kernels ----------------------------------

// pooled[b,c,h] = mean_w x_skel[b,c,h,w]
__global__ __launch_bounds__(256) void pool_kernel(const float* __restrict__ xs,
                                                   float* __restrict__ pooled) {
  int idx = blockIdx.x * blockDim.x + threadIdx.x;
  if (idx >= NB * CS * NS) return;
  const float* p = xs + (size_t)idx * NS;   // w is innermost, contiguous
  float s = 0.f;
#pragma unroll
  for (int w = 0; w < NS; ++w) s += p[w];
  pooled[idx] = s * (1.0f / 25.0f);
}

// kmat[b,i,s] = bk[i] + sum_c Wk[i,c] * pooled[b,c,s]   (f32)
__global__ __launch_bounds__(256) void k_kernel(const float* __restrict__ Wk,
                                                const float* __restrict__ bk,
                                                const float* __restrict__ pooled,
                                                float* __restrict__ kmat) {
  int idx = blockIdx.x * blockDim.x + threadIdx.x;
  if (idx >= NB * CI * NS) return;
  int s = idx % NS, i = (idx / NS) % CI, b = idx / (NS * CI);
  float acc = bk[i];
  const float* wk = Wk + (size_t)i * CS;
  const float* pl = pooled + (size_t)b * CS * NS + s;
  for (int c = 0; c < CS; ++c) acc += wk[c] * pl[(size_t)c * NS];
  kmat[idx] = acc;
}

// vws[b,r,s] = bf16(bv[r] + sum_c Wv[r,c]*pooled[b,c,s]), zero-padded s>=25
__global__ __launch_bounds__(256) void v_kernel(const float* __restrict__ Wv,
                                                const float* __restrict__ bv,
                                                const float* __restrict__ pooled,
                                                unsigned short* __restrict__ vws) {
  int idx = blockIdx.x * blockDim.x + threadIdx.x;
  if (idx >= NB * CR * NSP) return;
  int s = idx % NSP, r = (idx / NSP) % CR, b = idx / (NSP * CR);
  unsigned short o = 0;
  if (s < NS) {
    float acc = bv[r];
    const float* wv = Wv + (size_t)r * CS;
    const float* pl = pooled + (size_t)b * CS * NS + s;
    for (int c = 0; c < CS; ++c) acc += wv[c] * pl[(size_t)c * NS];
    o = f2bf(acc);
  }
  vws[idx] = o;
}

// mws[b,s,c] = bf16(sum_i Wq[i,c]*kmat[b,i,s])  (s-major for B-fragment reads)
__global__ __launch_bounds__(256) void m_kernel(const float* __restrict__ Wq,
                                                const float* __restrict__ kmat,
                                                unsigned short* __restrict__ mws) {
  int idx = blockIdx.x * blockDim.x + threadIdx.x;
  if (idx >= NB * NSP * CR) return;
  int c = idx % CR, s = (idx / CR) % NSP, b = idx / (CR * NSP);
  unsigned short o = 0;
  if (s < NS) {
    float acc = 0.f;
    const float* km = kmat + ((size_t)b * CI) * NS + s;
    for (int i = 0; i < CI; ++i) acc += Wq[(size_t)i * CR + c] * km[(size_t)i * NS];
    o = f2bf(acc);
  }
  mws[idx] = o;
}

// e0[b,s] = sum_i bq[i]*kmat[b,i,s]
__global__ __launch_bounds__(256) void e0_kernel(const float* __restrict__ bq,
                                                 const float* __restrict__ kmat,
                                                 float* __restrict__ e0) {
  int idx = blockIdx.x * blockDim.x + threadIdx.x;
  if (idx >= NB * NSP) return;
  int s = idx % NSP, b = idx / NSP;
  float acc = 0.f;
  if (s < NS) {
    const float* km = kmat + ((size_t)b * CI) * NS + s;
    for (int i = 0; i < CI; ++i) acc += bq[i] * km[(size_t)i * NS];
  }
  e0[idx] = acc;
}

// --------------------------- fused attention -------------------------------
// grid (49 hw-tiles, 16 batches), 128 threads = 4 wave32.
__global__ __launch_bounds__(128) void fused_attn(
    const float* __restrict__ xrgb,          // [16][1024][784] f32
    const float* __restrict__ e0ws,          // [16][32] f32
    const unsigned short* __restrict__ vws,  // [16][1024][32] bf16
    const unsigned short* __restrict__ mws,  // [16][32][1024] bf16
    const float* __restrict__ gamma,
    float* __restrict__ out) {
  __shared__ __attribute__((aligned(16))) unsigned short x_lds[4][16 * VROW];
  __shared__ __attribute__((aligned(16))) unsigned short a_lds[16 * VROW];
  __shared__ __attribute__((aligned(16))) float e_lds[4][16][NSP];

  const int tid  = threadIdx.x;
  const int w    = tid >> 5;
  const int lane = tid & 31;
  const int half = lane >> 4;   // 0: lanes 0-15, 1: lanes 16-31
  const int l15  = lane & 15;
  const int b    = blockIdx.y;
  const int hw0  = blockIdx.x * 16;

  // ---- Phase 1: energy[16hw x 32s] = x^T * M, K=1024 split 8 steps/wave ---
  f32x8 eacc0 = {}; f32x8 eacc1 = {};
  for (int step = 0; step < 8; ++step) {
    const int c0 = (w * 8 + step) * 32;
    {
      // lane loads one c-row of 16 consecutive hw (64B coalesced), transposes
      // into bf16 LDS tile [16hw][32c] (wave-private buffer).
      const float* src = xrgb + ((size_t)b * CR + (c0 + lane)) * HW + hw0;
      float4 f0 = reinterpret_cast<const float4*>(src)[0];
      float4 f1 = reinterpret_cast<const float4*>(src)[1];
      float4 f2 = reinterpret_cast<const float4*>(src)[2];
      float4 f3 = reinterpret_cast<const float4*>(src)[3];
      unsigned short* dst = &x_lds[w][lane];
      dst[0*VROW]  = f2bf(f0.x); dst[1*VROW]  = f2bf(f0.y);
      dst[2*VROW]  = f2bf(f0.z); dst[3*VROW]  = f2bf(f0.w);
      dst[4*VROW]  = f2bf(f1.x); dst[5*VROW]  = f2bf(f1.y);
      dst[6*VROW]  = f2bf(f1.z); dst[7*VROW]  = f2bf(f1.w);
      dst[8*VROW]  = f2bf(f2.x); dst[9*VROW]  = f2bf(f2.y);
      dst[10*VROW] = f2bf(f2.z); dst[11*VROW] = f2bf(f2.w);
      dst[12*VROW] = f2bf(f3.x); dst[13*VROW] = f2bf(f3.y);
      dst[14*VROW] = f2bf(f3.z); dst[15*VROW] = f2bf(f3.w);
    }
    __syncthreads();
    // A frag: row hw = l15; K runs {0-15} / {16-31} per half
    const unsigned short* xa = &x_lds[w][l15 * VROW + (half << 4)];
    bf16x16 afrag = load_frag16(xa, xa + 8);
    // B frags straight from global (L2-resident, zero reuse within block):
    // col s = l15 (+16), 16 consecutive K(c) values per lane
    const unsigned short* mb =
        mws + ((size_t)(b * NSP + l15)) * CR + c0 + (half << 4);
    bf16x16 bf0 = load_frag16(mb, mb + 8);
    bf16x16 bf1 = load_frag16(mb + 16 * CR, mb + 16 * CR + 8);
    eacc0 = __builtin_amdgcn_wmma_f32_16x16x32_bf16(false, afrag, false, bf0,
                                                    (short)0, eacc0, false, false);
    eacc1 = __builtin_amdgcn_wmma_f32_16x16x32_bf16(false, afrag, false, bf1,
                                                    (short)0, eacc1, false, false);
  }

  // ---- Phase 2: reduce 4 wave-partials + softmax over s<25 ----------------
#pragma unroll
  for (int j = 0; j < 8; ++j) {            // C layout: row = j + 8*half, col = l15
    e_lds[w][j + (half << 3)][l15]      = eacc0[j];
    e_lds[w][j + (half << 3)][l15 + 16] = eacc1[j];
  }
  __syncthreads();

  if (tid < 16) {
    float ev[NS];
    float m = -1e30f;
#pragma unroll
    for (int s = 0; s < NS; ++s) {
      float e = e_lds[0][tid][s] + e_lds[1][tid][s] +
                e_lds[2][tid][s] + e_lds[3][tid][s] + e0ws[b * NSP + s];
      ev[s] = e;
      m = fmaxf(m, e);
    }
    float sum = 0.f;
#pragma unroll
    for (int s = 0; s < NS; ++s) { float p = __expf(ev[s] - m); ev[s] = p; sum += p; }
    float inv = 1.0f / sum;
    unsigned short* ar = &a_lds[tid * VROW];   // attn stored [hw][s] (B layout)
#pragma unroll
    for (int s = 0; s < NS; ++s) ar[s] = f2bf(ev[s] * inv);
#pragma unroll
    for (int s = NS; s < NSP; ++s) ar[s] = 0;
  }
  __syncthreads();

  // ---- Phase 3: out[r,hw] = gamma * (v * attn^T) + x_rgb, 16 r-tiles/wave -
  const float g = gamma[0];
  for (int t = w; t < CR / 16; t += 4) {
    const int r0 = t * 16;
    // A frag from global v (L2-resident): row r = r0+l15, 16 consecutive K(s) per half
    const unsigned short* va =
        vws + ((size_t)b * CR + r0 + l15) * NSP + (half << 4);
    bf16x16 afrag = load_frag16(va, va + 8);
    // B frag from LDS attn: col hw = l15, 16 consecutive K(s)
    const unsigned short* ab = &a_lds[l15 * VROW + (half << 4)];
    bf16x16 bfrag = load_frag16(ab, ab + 8);
    f32x8 acc = {};
    acc = __builtin_amdgcn_wmma_f32_16x16x32_bf16(false, afrag, false, bfrag,
                                                  (short)0, acc, false, false);
    const int rbase = r0 + (half << 3);
#pragma unroll
    for (int j = 0; j < 8; ++j) {
      const size_t idx = ((size_t)b * CR + rbase + j) * HW + hw0 + l15;
      out[idx] = g * acc[j] + xrgb[idx];
    }
  }
}

// --------------------------- launcher --------------------------------------
extern "C" void kernel_launch(void* const* d_in, const int* in_sizes, int n_in,
                              void* d_out, int out_size, void* d_ws, size_t ws_size,
                              hipStream_t stream) {
  (void)in_sizes; (void)n_in; (void)out_size; (void)ws_size;
  const float* x_rgb  = (const float*)d_in[0];
  const float* x_skel = (const float*)d_in[1];
  const float* Wq     = (const float*)d_in[2];
  const float* bq     = (const float*)d_in[3];
  const float* Wk     = (const float*)d_in[4];
  const float* bk     = (const float*)d_in[5];
  const float* Wv     = (const float*)d_in[6];
  const float* bv     = (const float*)d_in[7];
  const float* gamma  = (const float*)d_in[8];
  float* out = (float*)d_out;

  // workspace layout (total ~3.33 MB)
  char* ws = (char*)d_ws;
  float* pooled        = (float*)(ws);                     // 16*256*25*4  = 409600
  float* kmat          = (float*)(ws + 409600);            // 16*512*25*4  = 819200
  float* e0            = (float*)(ws + 1228800);           // 16*32*4      = 2048
  unsigned short* vws  = (unsigned short*)(ws + 1230848);  // 16*1024*32*2 = 1048576
  unsigned short* mws  = (unsigned short*)(ws + 2279424);  // 16*32*1024*2 = 1048576

  pool_kernel<<<(NB * CS * NS + 255) / 256, 256, 0, stream>>>(x_skel, pooled);
  k_kernel   <<<(NB * CI * NS + 255) / 256, 256, 0, stream>>>(Wk, bk, pooled, kmat);
  v_kernel   <<<(NB * CR * NSP + 255) / 256, 256, 0, stream>>>(Wv, bv, pooled, vws);
  m_kernel   <<<(NB * NSP * CR + 255) / 256, 256, 0, stream>>>(Wq, kmat, mws);
  e0_kernel  <<<(NB * NSP + 255) / 256, 256, 0, stream>>>(bq, kmat, e0);
  fused_attn <<<dim3(HW / 16, NB), 128, 0, stream>>>(x_rgb, e0, vws, mws, gamma, out);
}
